// myPCANorm_noRec_38560216383957
// MI455X (gfx1250) — matrix operand order repaired and checked by reading, not compiled, not run on hardware
//
#include <hip/hip_runtime.h>

// ---------------- problem constants ----------------
#define C_DIM   64
#define HW      16384           // 128*128
#define N_IMG   64
#define M_TOT   (N_IMG * HW)    // 1,048,576
#define EPS_F   1e-5f
#define NPI     10
#define K_EIG   32

// ---------------- workspace layout (floats) ----------------
#define WS_GRAM 0        // 4096  (atomically accumulated, zeroed each call)
#define WS_SUM  4096     // 64
#define WS_MU   4160     // 64
#define WS_COV  4224     // 4096
#define WS_WF   8320     // 4096  (fused whitening matrix, f32)
#define WS_OFF  12416    // 64
#define WS_ZERO_COUNT 4160

// ---------------- vector types for WMMA ----------------
typedef __attribute__((ext_vector_type(16))) __bf16         v16bf;
typedef __attribute__((ext_vector_type(8)))  float          v8f;
typedef __attribute__((ext_vector_type(8)))  unsigned short u16x8;
typedef __attribute__((ext_vector_type(16))) unsigned short u16x16;

// round-to-nearest-even f32 -> bf16 (bit pattern as ushort)
__device__ __forceinline__ unsigned short f2bf(float f) {
  unsigned u = __builtin_bit_cast(unsigned, f);
  u += 0x7FFFu + ((u >> 16) & 1u);
  return (unsigned short)(u >> 16);
}

// combine two 8x u16 LDS vectors into one 16x bf16 WMMA fragment
__device__ __forceinline__ v16bf make_frag(const unsigned short* p0,
                                           const unsigned short* p1) {
  u16x8 lo = *(const u16x8*)p0;
  u16x8 hi = *(const u16x8*)p1;
  u16x16 u = __builtin_shufflevector(lo, hi,
      0, 1, 2, 3, 4, 5, 6, 7, 8, 9, 10, 11, 12, 13, 14, 15);
  return __builtin_bit_cast(v16bf, u);
}

__device__ __forceinline__ v8f wmma_bf16(v16bf a, v16bf b, v8f c) {
  return __builtin_amdgcn_wmma_f32_16x16x32_bf16(
      /*neg_a=*/false, a, /*neg_b=*/false, b,
      /*c_mod=*/(short)0, c, /*reuse_a=*/false, /*reuse_b=*/false);
}

// ============================================================
// Kernel 0: zero the atomic accumulators (ws is poisoned 0xAA)
// ============================================================
__global__ void zero_ws_kernel(float* ws) {
  int i = blockIdx.x * blockDim.x + threadIdx.x;
  if (i < WS_ZERO_COUNT) ws[i] = 0.0f;
}

// ============================================================
// Kernel 1: Gram matrix  gram += X * X^T  (+ per-channel sums)
// 512 blocks (64 images x 8 slabs of 2048 cols), 256 thr = 8 waves.
// 64-column chunks, double-buffered LDS -> ONE barrier per step,
// next chunk's global loads issued before the WMMAs.
// ============================================================
#define GSTRIDE 72   // u16 per LDS row (128 B data + 16 B pad, 16 B aligned)
#define GSTEPS  32   // 2048 / 64

__global__ __launch_bounds__(256) void gram_kernel(
    const float* __restrict__ x, float* __restrict__ ws) {
  __shared__ __attribute__((aligned(16))) unsigned short xt[2][C_DIM * GSTRIDE];
  __shared__ float sumbuf[C_DIM];

  const int tid  = threadIdx.x;
  const int lane = tid & 31;
  const int wave = tid >> 5;

  const int blk = blockIdx.x;
  const int n   = blk >> 3;        // image
  const int seg = blk & 7;         // 2048-column slab

  // loader mapping: thread -> (channel, 16 consecutive m-columns)
  const int c_ld = tid >> 2;
  const int cb   = (tid & 3) * 16;
  const float* xbase =
      x + (size_t)n * C_DIM * HW + (size_t)c_ld * HW + seg * 2048 + cb;

  // tile assignment: wave w -> row tile ti = w>>1, col tiles (w&1)*2, +1
  const int ti  = wave >> 1;
  const int tj0 = (wave & 1) * 2;

  v8f acc0 = {0.f, 0.f, 0.f, 0.f, 0.f, 0.f, 0.f, 0.f};
  v8f acc1 = {0.f, 0.f, 0.f, 0.f, 0.f, 0.f, 0.f, 0.f};
  float sacc = 0.f;
  if (tid < C_DIM) sumbuf[tid] = 0.f;

  // fragment addressing per ISA 16-bit A/B layouts (wave32)
  const int arow  = 16 * ti + (lane & 15);
  const int khalf = (lane >> 4) * 8;        // A: K pairs split low/high lanes
  const int brow0 = 16 * tj0 + (lane & 15);
  const int brow1 = brow0 + 16;
  const int kb    = (lane >> 4) * 16;       // unused; B frags use khalf, same k order as A

  auto stage = [&](int buf, float4 q0, float4 q1, float4 q2, float4 q3) {
    u16x8 v0{}, v1{};
    v0[0] = f2bf(q0.x); v0[1] = f2bf(q0.y); v0[2] = f2bf(q0.z); v0[3] = f2bf(q0.w);
    v0[4] = f2bf(q1.x); v0[5] = f2bf(q1.y); v0[6] = f2bf(q1.z); v0[7] = f2bf(q1.w);
    v1[0] = f2bf(q2.x); v1[1] = f2bf(q2.y); v1[2] = f2bf(q2.z); v1[3] = f2bf(q2.w);
    v1[4] = f2bf(q3.x); v1[5] = f2bf(q3.y); v1[6] = f2bf(q3.z); v1[7] = f2bf(q3.w);
    *(u16x8*)&xt[buf][c_ld * GSTRIDE + cb]     = v0;
    *(u16x8*)&xt[buf][c_ld * GSTRIDE + cb + 8] = v1;
    sacc += ((q0.x + q0.y) + (q0.z + q0.w)) + ((q1.x + q1.y) + (q1.z + q1.w)) +
            ((q2.x + q2.y) + (q2.z + q2.w)) + ((q3.x + q3.y) + (q3.z + q3.w));
  };

  // ---- prologue: stage chunk 0 ----
  {
    const float* gp = xbase;
    stage(0, *(const float4*)(gp),     *(const float4*)(gp + 4),
             *(const float4*)(gp + 8), *(const float4*)(gp + 12));
  }
  __syncthreads();

  for (int step = 0; step < GSTEPS; ++step) {
    const int  cur  = step & 1;
    const bool more = (step + 1 < GSTEPS);

    // issue next chunk's global loads before computing (latency overlap)
    float4 q0, q1, q2, q3;
    if (more) {
      const float* gp = xbase + (step + 1) * 64;
      q0 = *(const float4*)(gp);
      q1 = *(const float4*)(gp + 4);
      q2 = *(const float4*)(gp + 8);
      q3 = *(const float4*)(gp + 12);
      __builtin_prefetch(gp + 64, 0, 3);
    }

    // 4 WMMAs on the current buffer (K = 64 in two 32-chunks)
    const unsigned short* base = &xt[cur][0];
#pragma unroll
    for (int kc = 0; kc < 2; ++kc) {
      const int ko = kc * 32;
      v16bf a  = make_frag(base + arow * GSTRIDE + ko + khalf,
                           base + arow * GSTRIDE + ko + 16 + khalf);
      v16bf b0 = make_frag(base + brow0 * GSTRIDE + ko + khalf,
                           base + brow0 * GSTRIDE + ko + 16 + khalf);
      v16bf b1 = make_frag(base + brow1 * GSTRIDE + ko + khalf,
                           base + brow1 * GSTRIDE + ko + 16 + khalf);
      acc0 = wmma_bf16(a, b0, acc0);
      acc1 = wmma_bf16(a, b1, acc1);
    }

    // stage next chunk into the other buffer; single barrier per step
    if (more) stage(1 - cur, q0, q1, q2, q3);
    __syncthreads();
  }

  // ---- channel sums: block-local LDS reduce, then one global atomic ----
  atomicAdd(&sumbuf[c_ld], sacc);
  __syncthreads();
  if (tid < C_DIM) atomicAdd(&ws[WS_SUM + tid], sumbuf[tid]);

  // ---- commit partial Gram tiles (C/D layout: lane=N, VGPR r -> M) ----
  const int rofs = 8 * (lane >> 4);
  const int ncol = lane & 15;
  float* gram = ws + WS_GRAM;
#pragma unroll
  for (int r = 0; r < 8; ++r) {
    int mrow = 16 * ti + r + rofs;
    atomicAdd(&gram[mrow * 64 + 16 * tj0       + ncol], acc0[r]);
    atomicAdd(&gram[mrow * 64 + 16 * (tj0 + 1) + ncol], acc1[r]);
  }
}

// ============================================================
// Kernel 2: cov = gram/M - mu*mu^T + eps*I ; also store mu
// ============================================================
__global__ __launch_bounds__(256) void finalize_cov_kernel(float* ws) {
  __shared__ float mu_s[C_DIM];
  const int tid = threadIdx.x;
  const float invM = 1.0f / (float)M_TOT;
  if (tid < C_DIM) {
    float m = ws[WS_SUM + tid] * invM;
    mu_s[tid] = m;
    ws[WS_MU + tid] = m;
  }
  __syncthreads();
  for (int idx = tid; idx < C_DIM * C_DIM; idx += 256) {
    int i = idx >> 6, j = idx & 63;
    float c = ws[WS_GRAM + idx] * invM - mu_s[i] * mu_s[j];
    if (i == j) c += EPS_F;
    ws[WS_COV + idx] = c;
  }
}

// ============================================================
// Kernel 3: power iteration + Hotelling deflation (1 block, 64 thr)
//   Wf = diag(weight) * vs^T * diag(1/sqrt(lam+eps)) * vs
//   off = bias - Wf * mu
// ============================================================
__global__ __launch_bounds__(64) void eig_kernel(
    const float* __restrict__ vinit, const float* __restrict__ weight,
    const float* __restrict__ bias, float* __restrict__ ws) {
  __shared__ float cov[C_DIM][C_DIM + 1];
  __shared__ float v[C_DIM], w[C_DIM], red[C_DIM], muv[C_DIM];
  __shared__ float vs[K_EIG][C_DIM];
  __shared__ float rs[K_EIG];

  const int c = threadIdx.x;  // 0..63
  for (int idx = c; idx < C_DIM * C_DIM; idx += C_DIM)
    cov[idx >> 6][idx & 63] = ws[WS_COV + idx];
  muv[c] = ws[WS_MU + c];
  __syncthreads();

  for (int k = 0; k < K_EIG; ++k) {
    v[c] = vinit[k * C_DIM + c];
    __syncthreads();

    for (int it = 0; it < NPI; ++it) {
      float acc = 0.f;
#pragma unroll 8
      for (int j = 0; j < C_DIM; ++j) acc += cov[c][j] * v[j];
      w[c] = acc;
      red[c] = acc * acc;
      __syncthreads();
      for (int s = 32; s > 0; s >>= 1) {
        if (c < s) red[c] += red[c + s];
        __syncthreads();
      }
      float inv = rsqrtf(red[0]);
      v[c] = w[c] * inv;
      __syncthreads();
    }

    // Rayleigh quotient
    float acc = 0.f;
#pragma unroll 8
    for (int j = 0; j < C_DIM; ++j) acc += cov[c][j] * v[j];
    red[c] = v[c] * acc;
    __syncthreads();
    for (int s = 32; s > 0; s >>= 1) {
      if (c < s) red[c] += red[c + s];
      __syncthreads();
    }
    float lam = red[0];
    __syncthreads();

    // deflate own row; record eigenpair
    float vc = v[c];
#pragma unroll 8
    for (int j = 0; j < C_DIM; ++j) cov[c][j] -= lam * vc * v[j];
    vs[k][c] = vc;
    if (c == 0) rs[k] = rsqrtf(lam + EPS_F);
    __syncthreads();
  }

  // fused whitening row for channel c
  const float wgt = weight[c];
  float off = bias[c];
  for (int j = 0; j < C_DIM; ++j) {
    float s = 0.f;
#pragma unroll 8
    for (int k = 0; k < K_EIG; ++k) s += vs[k][c] * vs[k][j] * rs[k];
    float wf = wgt * s;
    ws[WS_WF + c * C_DIM + j] = wf;
    off -= wf * muv[j];
  }
  ws[WS_OFF + c] = off;
}

// ============================================================
// Kernel 4: y = Wf * x + off   (K = channel dim = 64 -> 2 WMMAs/tile)
// A-frags built once per wave and held in VGPRs. 64-column chunks,
// double-buffered transposed LDS tile [m][ch]; one barrier per step.
// ============================================================
#define ASTRIDE 72   // u16 per LDS row (128 B data + 16 B pad)
#define ASTEPS  32   // 2048 / 64

__global__ __launch_bounds__(256) void apply_kernel(
    const float* __restrict__ x, const float* __restrict__ ws,
    float* __restrict__ out) {
  __shared__ __attribute__((aligned(16))) unsigned short wt[C_DIM * ASTRIDE];
  __shared__ __attribute__((aligned(16))) unsigned short xt[2][C_DIM * ASTRIDE];
  __shared__ float offs[C_DIM];

  const int tid  = threadIdx.x;
  const int lane = tid & 31;
  const int wave = tid >> 5;

  // ---- stage Wf as bf16 in LDS ----
  {
    const int r  = tid >> 2;
    const int c4 = (tid & 3) * 16;
    u16x8 v0{}, v1{};
#pragma unroll
    for (int q = 0; q < 8; ++q) {
      v0[q] = f2bf(ws[WS_WF + r * C_DIM + c4 + q]);
      v1[q] = f2bf(ws[WS_WF + r * C_DIM + c4 + 8 + q]);
    }
    *(u16x8*)&wt[r * ASTRIDE + c4]     = v0;
    *(u16x8*)&wt[r * ASTRIDE + c4 + 8] = v1;
  }
  if (tid < C_DIM) offs[tid] = ws[WS_OFF + tid];
  __syncthreads();

  const int ti  = wave >> 1;        // output channel tile (0..3)
  const int jmb = (wave & 1) * 2;   // first of two m sub-tiles (0 or 2)

  // A fragments for this wave's 16 output channels, K = 64 (two chunks)
  const int arow  = 16 * ti + (lane & 15);
  const int khalf = (lane >> 4) * 8;
  const v16bf a0 = make_frag(&wt[arow * ASTRIDE + khalf],
                             &wt[arow * ASTRIDE + 16 + khalf]);
  const v16bf a1 = make_frag(&wt[arow * ASTRIDE + 32 + khalf],
                             &wt[arow * ASTRIDE + 48 + khalf]);

  const int rofs = 8 * (lane >> 4);
  float oarr[8];
#pragma unroll
  for (int r = 0; r < 8; ++r) oarr[r] = offs[16 * ti + r + rofs];

  const int blk = blockIdx.x;       // 512 blocks
  const int n   = blk >> 3;
  const int seg = blk & 7;
  const int c_ld = tid >> 2;
  const int cb   = (tid & 3) * 16;  // local m-columns this thread loads
  const float* xbase =
      x + (size_t)n * C_DIM * HW + (size_t)c_ld * HW + seg * 2048 + cb;
  float* obase = out + (size_t)n * C_DIM * HW + seg * 2048;

  const int brow0 = jmb * 16 + (lane & 15);  // B: lane -> m (row of xt)
  const int brow1 = brow0 + 16;
  const int kbB   = (lane >> 4) * 16;        // B: channel split in K-chunk

  // scatter-transpose stage: xt[m][channel]
  auto stage = [&](int buf, float4 q0, float4 q1, float4 q2, float4 q3) {
    unsigned short* p = &xt[buf][0];
    p[(cb +  0) * ASTRIDE + c_ld] = f2bf(q0.x);
    p[(cb +  1) * ASTRIDE + c_ld] = f2bf(q0.y);
    p[(cb +  2) * ASTRIDE + c_ld] = f2bf(q0.z);
    p[(cb +  3) * ASTRIDE + c_ld] = f2bf(q0.w);
    p[(cb +  4) * ASTRIDE + c_ld] = f2bf(q1.x);
    p[(cb +  5) * ASTRIDE + c_ld] = f2bf(q1.y);
    p[(cb +  6) * ASTRIDE + c_ld] = f2bf(q1.z);
    p[(cb +  7) * ASTRIDE + c_ld] = f2bf(q1.w);
    p[(cb +  8) * ASTRIDE + c_ld] = f2bf(q2.x);
    p[(cb +  9) * ASTRIDE + c_ld] = f2bf(q2.y);
    p[(cb + 10) * ASTRIDE + c_ld] = f2bf(q2.z);
    p[(cb + 11) * ASTRIDE + c_ld] = f2bf(q2.w);
    p[(cb + 12) * ASTRIDE + c_ld] = f2bf(q3.x);
    p[(cb + 13) * ASTRIDE + c_ld] = f2bf(q3.y);
    p[(cb + 14) * ASTRIDE + c_ld] = f2bf(q3.z);
    p[(cb + 15) * ASTRIDE + c_ld] = f2bf(q3.w);
  };

  // ---- prologue: stage chunk 0 ----
  {
    const float* gp = xbase;
    stage(0, *(const float4*)(gp),     *(const float4*)(gp + 4),
             *(const float4*)(gp + 8), *(const float4*)(gp + 12));
  }
  __syncthreads();

  for (int step = 0; step < ASTEPS; ++step) {
    const int  cur  = step & 1;
    const bool more = (step + 1 < ASTEPS);

    float4 q0, q1, q2, q3;
    if (more) {
      const float* gp = xbase + (step + 1) * 64;
      q0 = *(const float4*)(gp);
      q1 = *(const float4*)(gp + 4);
      q2 = *(const float4*)(gp + 8);
      q3 = *(const float4*)(gp + 12);
      __builtin_prefetch(gp + 64, 0, 3);
    }

    // two 16x16 output tiles per wave, K = 64
    const unsigned short* base = &xt[cur][0];
    v8f acc0 = {0.f, 0.f, 0.f, 0.f, 0.f, 0.f, 0.f, 0.f};
    v8f acc1 = {0.f, 0.f, 0.f, 0.f, 0.f, 0.f, 0.f, 0.f};
#pragma unroll
    for (int kc = 0; kc < 2; ++kc) {
      const int ko = kc * 32;
      v16bf a  = (kc == 0) ? a0 : a1;
      v16bf b0 = make_frag(base + brow0 * ASTRIDE + ko + khalf,
                           base + brow0 * ASTRIDE + ko + 16 + khalf);
      v16bf b1 = make_frag(base + brow1 * ASTRIDE + ko + khalf,
                           base + brow1 * ASTRIDE + ko + 16 + khalf);
      acc0 = wmma_bf16(a, b0, acc0);
      acc1 = wmma_bf16(a, b1, acc1);
    }

    // write y[n][ch][m] = acc + off[ch]
    const int mcol0 = step * 64 + jmb * 16 + (lane & 15);
#pragma unroll
    for (int r = 0; r < 8; ++r) {
      int ch = 16 * ti + r + rofs;
      obase[(size_t)ch * HW + mcol0]      = acc0[r] + oarr[r];
      obase[(size_t)ch * HW + mcol0 + 16] = acc1[r] + oarr[r];
    }

    if (more) stage(1 - cur, q0, q1, q2, q3);
    __syncthreads();
  }
}

// ============================================================
extern "C" void kernel_launch(void* const* d_in, const int* in_sizes, int n_in,
                              void* d_out, int out_size, void* d_ws,
                              size_t ws_size, hipStream_t stream) {
  const float* x      = (const float*)d_in[0];  // [64,64,128,128]
  const float* weight = (const float*)d_in[1];  // [64,1]
  const float* bias   = (const float*)d_in[2];  // [64,1]
  const float* vinit  = (const float*)d_in[3];  // [32,64]
  float* out = (float*)d_out;
  float* ws  = (float*)d_ws;

  zero_ws_kernel<<<dim3((WS_ZERO_COUNT + 255) / 256), dim3(256), 0, stream>>>(ws);
  gram_kernel<<<dim3(512), dim3(256), 0, stream>>>(x, ws);
  finalize_cov_kernel<<<dim3(1), dim3(256), 0, stream>>>(ws);
  eig_kernel<<<dim3(1), dim3(64), 0, stream>>>(vinit, weight, bias, ws);
  apply_kernel<<<dim3(512), dim3(256), 0, stream>>>(x, ws, out);
}
